// GlobalAttention_45526653337807
// MI455X (gfx1250) — matrix operand order repaired and checked by reading, not compiled, not run on hardware
//
#include <hip/hip_runtime.h>
#include <hip/hip_bf16.h>

typedef __attribute__((ext_vector_type(16))) _Float16 v16h;
typedef __attribute__((ext_vector_type(8)))  _Float16 v8h;
typedef __attribute__((ext_vector_type(8)))  float    v8f;

#define BB 2
#define CC 64
#define DD 8
#define NN 9216
#define KPAD 32                                 // zero-padded K row (halves)
#define LOG2E 1.44269504088896f

#if __has_builtin(__builtin_amdgcn_exp2f)
#define EXP2F __builtin_amdgcn_exp2f
#else
#define EXP2F exp2f
#endif

#if __has_builtin(__builtin_amdgcn_cvt_pkrtz)
static __device__ __forceinline__ unsigned pk2(float a, float b) {
    return __builtin_bit_cast(unsigned, __builtin_amdgcn_cvt_pkrtz(a, b));
}
#else
static __device__ __forceinline__ unsigned pk2(float a, float b) {
    union { _Float16 h[2]; unsigned u; } x;
    x.h[0] = (_Float16)a; x.h[1] = (_Float16)b;
    return x.u;
}
#endif

// ---------------------------------------------------------------------------
// Prologue: per (b, n) position, compute q[8] (pre-scaled by log2e, stored
// (B,N,8)), k[8] (stored zero-padded (B,N,32) so the WMMA A-operand is one
// unconditional contiguous load on every lane), v[64] (stored (B,C,N)).
// ---------------------------------------------------------------------------
__global__ __launch_bounds__(256) void qkv_kernel(
    const float* __restrict__ x,
    const float* __restrict__ wq, const float* __restrict__ bq,
    const float* __restrict__ wk, const float* __restrict__ bk,
    const float* __restrict__ wv, const float* __restrict__ bv,
    _Float16* __restrict__ qt, _Float16* __restrict__ ktz,
    _Float16* __restrict__ vt)
{
    int gid = blockIdx.x * blockDim.x + threadIdx.x;   // 0 .. B*N-1
    int b = gid / NN, n = gid - b * NN;
    const float* xb = x + (size_t)b * CC * NN + n;
    float xs[CC];
    #pragma unroll
    for (int c = 0; c < CC; ++c) xs[c] = xb[(size_t)c * NN];

    _Float16* kr = ktz + ((size_t)b * NN + n) * KPAD;
    for (int o = 0; o < DD; ++o) {
        float aq = bq[o], ak = bk[o];
        #pragma unroll
        for (int c = 0; c < CC; ++c) {
            aq += wq[o * CC + c] * xs[c];
            ak += wk[o * CC + c] * xs[c];
        }
        // q pre-scaled into log2 domain: WMMA scores come out ready for exp2
        qt[((size_t)b * NN + n) * DD + o] = (_Float16)(aq * LOG2E);
        kr[o] = (_Float16)ak;
    }
    #pragma unroll
    for (int o = DD; o < KPAD; ++o) kr[o] = (_Float16)0.0f;

    for (int o = 0; o < CC; ++o) {
        float a = bv[o];
        #pragma unroll
        for (int c = 0; c < CC; ++c) a += wv[o * CC + c] * xs[c];
        vt[((size_t)b * CC + o) * NN + n] = (_Float16)a;
    }
}

// ---------------------------------------------------------------------------
// Flash attention: one wave per 16-query tile; stream keys in tiles of 64.
// S^T = K_tile x Q (softmax dim lane-local + one shfl_xor(16) for the max).
// PV reduction uses a lane-local slot<->key permutation so the P operand is
// built with zero cross-lane traffic and V loads are contiguous.
// Denominator accumulated on the matrix pipe via a ones-row A operand.
// ---------------------------------------------------------------------------
__global__ __launch_bounds__(128) void attn_kernel(
    const float* __restrict__ x,
    const _Float16* __restrict__ qt, const _Float16* __restrict__ ktz,
    const _Float16* __restrict__ vt, const float* __restrict__ gamma,
    float* __restrict__ out)
{
    const int lane = threadIdx.x & 31;
    const int ln   = lane & 15;
    const int hi   = lane >> 4;                 // partner-half flag
    const int wave = blockIdx.x * (blockDim.x >> 5) + (threadIdx.x >> 5);
    const int tilesPerB = NN / 16;              // 576
    const int b  = wave / tilesPerB;
    const int i0 = (wave - b * tilesPerB) * 16;

    // Q operand (B-layout): lanes 0-15 hold dims 0..7 in halves 0..7; rest 0.
    v16h qop = {};
    if (!hi) {
        v8h qv = *(const v8h*)(qt + ((size_t)b * NN + i0 + ln) * DD);
        #pragma unroll
        for (int t = 0; t < 8; ++t) qop[t] = qv[t];
    }

    // Ones A-operand: row M=0 all-ones (covers all K slots; permutation-safe)
    v16h ones = {};
    if (ln == 0) {
        #pragma unroll
        for (int t = 0; t < 16; ++t) ones[t] = (_Float16)1.0f;
    }

    v8f acc[4] = {{}, {}, {}, {}};
    v8f accS = {};                              // row 0 = softmax denominator
    float m2 = -3.0e38f;                        // running max (log2 domain)

    const _Float16* kb = ktz + (size_t)b * NN * KPAD;
    const _Float16* vb = vt + (size_t)b * CC * NN;

    for (int j0 = 0; j0 < NN; j0 += 64) {
        // --- K operands: one unconditional contiguous 32B load per 16 keys.
        //     hi=0 lanes hit dims 0..7 + zero pad; hi=1 lanes land fully in
        //     the zero pad (their A-layout slots are dims 8..15/24..31 == 0).
        v16h kop[4];
        #pragma unroll
        for (int g = 0; g < 4; ++g)
            kop[g] = *(const v16h*)(kb + (size_t)(j0 + g * 16 + ln) * KPAD + hi * 8);

        v8f z = {};
        v8f s[4];
        #pragma unroll
        for (int g = 0; g < 4; ++g)
            s[g] = __builtin_amdgcn_wmma_f32_16x16x32_f16(false, kop[g], false, qop,
                                                          (short)0, z, false, false);

        // --- tile max over this lane's 32 scores + partner lane ---
        float tmax = -3.0e38f;
        #pragma unroll
        for (int g = 0; g < 4; ++g)
            #pragma unroll
            for (int r = 0; r < 8; ++r) tmax = fmaxf(tmax, s[g][r]);
        tmax = fmaxf(tmax, __shfl_xor(tmax, 16, 32));

        // --- rescale only when some query's max actually grew (rare) ---
        if (__any(tmax > m2)) {
            float m2new = fmaxf(m2, tmax);
            float alpha = EXP2F(m2 - m2new);
            m2 = m2new;
            #pragma unroll
            for (int r = 0; r < 8; ++r) {
                acc[0][r] *= alpha; acc[1][r] *= alpha;
                acc[2][r] *= alpha; acc[3][r] *= alpha;
                accS[r]   *= alpha;
            }
        }

        // --- p = exp2(s - m); packed straight into B operands (lane-local:
        //     slot<->key permutation makes pop identical on both lane halves) ---
        union V16 { v16h v; unsigned u[8]; };
        V16 p01, p23;
        #pragma unroll
        for (int w = 0; w < 4; ++w) {
            p01.u[w]     = pk2(EXP2F(s[0][2*w] - m2), EXP2F(s[0][2*w+1] - m2));
            p01.u[w + 4] = pk2(EXP2F(s[1][2*w] - m2), EXP2F(s[1][2*w+1] - m2));
            p23.u[w]     = pk2(EXP2F(s[2][2*w] - m2), EXP2F(s[2][2*w+1] - m2));
            p23.u[w + 4] = pk2(EXP2F(s[3][2*w] - m2), EXP2F(s[3][2*w+1] - m2));
        }

        // --- denominator on the matrix pipe ---
        accS = __builtin_amdgcn_wmma_f32_16x16x32_f16(false, ones, false, p01.v,
                                                      (short)0, accS, false, false);
        accS = __builtin_amdgcn_wmma_f32_16x16x32_f16(false, ones, false, p23.v,
                                                      (short)0, accS, false, false);

        // --- PV: out(64x16) += V(64x64) x P(64x16); V loads contiguous under
        //     the same slot<->key permutation: lane reads v[c, j0+hi*8 ...] ---
        #pragma unroll
        for (int cb = 0; cb < 4; ++cb) {
            const _Float16* vp = vb + (size_t)(cb * 16 + ln) * NN + j0 + hi * 8;
            v8h va0 = *(const v8h*)(vp);
            v8h va1 = *(const v8h*)(vp + 16);
            v8h va2 = *(const v8h*)(vp + 32);
            v8h va3 = *(const v8h*)(vp + 48);
            v16h vop01, vop23;
            #pragma unroll
            for (int t = 0; t < 8; ++t) {
                vop01[t] = va0[t]; vop01[t + 8] = va1[t];
                vop23[t] = va2[t]; vop23[t + 8] = va3[t];
            }
            acc[cb] = __builtin_amdgcn_wmma_f32_16x16x32_f16(
                false, vop01, false, p01.v, (short)0, acc[cb], false, false);
            acc[cb] = __builtin_amdgcn_wmma_f32_16x16x32_f16(
                false, vop23, false, p23.v, (short)0, acc[cb], false, false);
        }
    }

    // --- epilogue: l lives in accS row 0 (lanes 0-15); broadcast, normalize ---
    float l = __shfl(accS[0], ln, 32);
    float invl = 1.0f / l;
    float g = gamma[0];
    #pragma unroll
    for (int cb = 0; cb < 4; ++cb) {
        #pragma unroll
        for (int r = 0; r < 8; ++r) {
            int c = cb * 16 + hi * 8 + r;
            size_t idx = ((size_t)b * CC + c) * NN + i0 + ln;
            out[idx] = g * (acc[cb][r] * invl) + x[idx];
        }
    }
}

// ---------------------------------------------------------------------------
extern "C" void kernel_launch(void* const* d_in, const int* in_sizes, int n_in,
                              void* d_out, int out_size, void* d_ws, size_t ws_size,
                              hipStream_t stream) {
    (void)in_sizes; (void)n_in; (void)out_size; (void)ws_size;
    const float* x     = (const float*)d_in[0];
    const float* wq    = (const float*)d_in[1];
    const float* bq    = (const float*)d_in[2];
    const float* wk    = (const float*)d_in[3];
    const float* bk    = (const float*)d_in[4];
    const float* wv    = (const float*)d_in[5];
    const float* bv    = (const float*)d_in[6];
    const float* gamma = (const float*)d_in[7];
    float* out = (float*)d_out;

    _Float16* qt  = (_Float16*)d_ws;                      // B*N*8   f16
    _Float16* ktz = qt + (size_t)BB * NN * DD;            // B*N*32  f16 (padded)
    _Float16* vt  = ktz + (size_t)BB * NN * KPAD;         // B*C*N   f16

    qkv_kernel<<<(BB * NN) / 256, 256, 0, stream>>>(x, wq, bq, wk, bk, wv, bv,
                                                    qt, ktz, vt);

    const int wavesTotal = BB * (NN / 16);                // 1152
    attn_kernel<<<wavesTotal / 4, 128, 0, stream>>>(x, qt, ktz, vt, gamma, out);
}
